// CustomDeepseekDBOMoE_31894427140772
// MI455X (gfx1250) — compile-verified, matrix operand order checked by reading, not run on hardware
//
#include <hip/hip_runtime.h>
#include <hip/hip_bf16.h>
#include <stdint.h>

#define T_TOK 4096
#define HID   2048
#define NEXP  32
#define INTER 1024
#define TOPK  8
#define NGRP  8
#define TOPKG 4
#define SIC   2048   // shared intermediate
#define AP    72     // LDS pitch (bf16 elems) for 64-wide tiles, 16B aligned rows

typedef __attribute__((ext_vector_type(16))) __bf16        v16bf;
typedef __attribute__((ext_vector_type(8)))  float         v8f;
typedef __attribute__((ext_vector_type(8)))  unsigned int  u32x8;
typedef __attribute__((ext_vector_type(4)))  unsigned int  u32x4;
typedef __attribute__((ext_vector_type(8)))  int           i32x8;
typedef __attribute__((ext_vector_type(4)))  int           i32x4;

__device__ __forceinline__ unsigned short f32_to_bf16(float f) {
  unsigned u = __builtin_bit_cast(unsigned, f);
  u += 0x7FFFu + ((u >> 16) & 1u);          // round-to-nearest-even
  return (unsigned short)(u >> 16);
}

// ---- Tensor Data Mover: 2-D tile (f32) global -> LDS ------------------------
__device__ __forceinline__ void tdm_load_2d(unsigned lds_off, const float* gptr,
                                            int tile_cols, int tile_rows,
                                            int stride_elems) {
  unsigned long long ga = (unsigned long long)(uintptr_t)gptr;
  u32x4 g0;
  g0.x = 1u;                                                    // count=1
  g0.y = lds_off;                                               // lds_addr
  g0.z = (unsigned)ga;                                          // global_addr[31:0]
  g0.w = (unsigned)((ga >> 32) & 0x01FFFFFFu) | (2u << 30);     // addr[56:32] | type=2
  unsigned td0 = (unsigned)stride_elems;                        // tensor_dim0
  unsigned td1 = (unsigned)tile_rows;                           // tensor_dim1
  i32x8 g1;
  g1[0] = (int)(2u << 16);                                      // data_size=4B, mask=0
  g1[1] = (int)((td0 & 0xFFFFu) << 16);                         // tensor_dim0[15:0]
  g1[2] = (int)((td0 >> 16) | ((td1 & 0xFFFFu) << 16));         // dim0 hi | dim1 lo
  g1[3] = (int)((td1 >> 16) | (((unsigned)tile_cols & 0xFFFFu) << 16)); // tile_dim0
  g1[4] = (int)((unsigned)tile_rows & 0xFFFFu);                 // tile_dim1
  g1[5] = (int)td0;                                             // dim0_stride lo32
  g1[6] = 0;
  g1[7] = 0;
  i32x4 z4 = {0, 0, 0, 0};
  i32x8 z8 = {0, 0, 0, 0, 0, 0, 0, 0};
  __builtin_amdgcn_tensor_load_to_lds(g0, g1, z4, z4, z8, 0);   // 6-arg variant
}

// ---- fragment load from LDS (A-style layout, 16 bf16 / lane) ---------------
__device__ __forceinline__ v16bf load_frag(const unsigned short* base, int pitch,
                                           int k0, int lane) {
  int mn = lane & 15, half = lane >> 4;
  const unsigned short* p = base + mn * pitch + k0 + half * 8;
  uint4 lo = *(const uint4*)p;        // K = k0 .. k0+7   (this half)
  uint4 hi = *(const uint4*)(p + 16); // K = k0+16 .. +23 (this half)
  u32x8 u;
  u[0] = lo.x; u[1] = lo.y; u[2] = lo.z; u[3] = lo.w;
  u[4] = hi.x; u[5] = hi.y; u[6] = hi.z; u[7] = hi.w;
  return __builtin_bit_cast(v16bf, u);
}

// ============================ kernel 0: cast ================================
__global__ void cast_bf16_kernel(const float* __restrict__ x,
                                 unsigned short* __restrict__ xb, int n) {
  int i = (blockIdx.x * blockDim.x + threadIdx.x) * 4;
  if (i >= n) return;
  float4 v = *(const float4*)(x + i);
  unsigned r0 = (unsigned)f32_to_bf16(v.x) | ((unsigned)f32_to_bf16(v.y) << 16);
  unsigned r1 = (unsigned)f32_to_bf16(v.z) | ((unsigned)f32_to_bf16(v.w) << 16);
  uint2 o; o.x = r0; o.y = r1;
  *(uint2*)(xb + i) = o;
}

// ============================ kernel 1: router ==============================
// one wave32 per token; lane == expert
__global__ __launch_bounds__(256)
void router_kernel(const float* __restrict__ x, const float* __restrict__ gw,
                   const float* __restrict__ gbias, float* __restrict__ topw,
                   int* __restrict__ cnt, int* __restrict__ rows) {
  int lane = threadIdx.x & 31;
  int wv   = threadIdx.x >> 5;
  int t    = blockIdx.x * 8 + wv;
  const float* xr = x + (size_t)t * HID;
  float acc = 0.f;
  #pragma unroll 4
  for (int h = 0; h < HID; h += 4) {
    acc += xr[h + 0] * gw[(h + 0) * NEXP + lane];
    acc += xr[h + 1] * gw[(h + 1) * NEXP + lane];
    acc += xr[h + 2] * gw[(h + 2) * NEXP + lane];
    acc += xr[h + 3] * gw[(h + 3) * NEXP + lane];
  }
  float s  = 1.f / (1.f + __expf(-acc));
  float sc = s + gbias[lane];

  // sum of top-2 within each group of 4 lanes
  int gbase = lane & ~3;
  float v0 = __shfl(sc, gbase + 0, 32);
  float v1 = __shfl(sc, gbase + 1, 32);
  float v2 = __shfl(sc, gbase + 2, 32);
  float v3 = __shfl(sc, gbase + 3, 32);
  float m1 = fmaxf(fmaxf(v0, v1), fmaxf(v2, v3));
  float r0 = v0, r1 = v1, r2 = v2, r3 = v3;
  if      (r0 == m1) r0 = -1e30f;
  else if (r1 == m1) r1 = -1e30f;
  else if (r2 == m1) r2 = -1e30f;
  else               r3 = -1e30f;
  float gs = m1 + fmaxf(fmaxf(r0, r1), fmaxf(r2, r3));

  // rank groups, keep top-4
  int g = lane >> 2, grank = 0;
  #pragma unroll
  for (int gg = 0; gg < NGRP; ++gg) {
    float og = __shfl(gs, gg * 4, 32);
    grank += (og > gs) || (og == gs && gg < g);
  }
  float masked = (grank < TOPKG) ? sc : -1e30f;

  // rank experts, keep top-8
  int rank = 0;
  #pragma unroll
  for (int l = 0; l < 32; ++l) {
    float ov = __shfl(masked, l, 32);
    rank += (ov > masked) || (ov == masked && l < lane);
  }
  bool sel = rank < TOPK;
  float ws = sel ? s : 0.f;
  #pragma unroll
  for (int o = 16; o; o >>= 1) ws += __shfl_xor(ws, o, 32);
  float w = s / (ws + 1e-20f);
  if (sel) {
    topw[t * TOPK + rank] = w;
    int pos = atomicAdd(&cnt[lane], 1);
    rows[(size_t)lane * T_TOK + pos] = (t << 3) | rank;
  }
}

// ======================= kernel 2: gemm1 + silu*up ==========================
__global__ __launch_bounds__(256)
void gemm1_silu_kernel(const unsigned short* __restrict__ xb,
                       const float* __restrict__ wgu,
                       const int* __restrict__ rows, const int* __restrict__ cnt,
                       unsigned short* __restrict__ inter,
                       int ic, int gathered, int rshift) {
  __shared__ int            s_rows[128];
  __shared__ unsigned short s_a[128 * AP];
  __shared__ float          s_wf[64 * 64];
  __shared__ unsigned short s_wb[2][64 * AP];

  int bx = blockIdx.x, icol0 = blockIdx.y * 64;
  int e = 0, m0, cnt_e = 0;
  const float* w = wgu;
  long ld2 = 2 * ic;
  if (gathered) {
    e = bx >> 5; m0 = (bx & 31) * 128;
    cnt_e = cnt[e];
    if (m0 >= cnt_e) return;
    w += (size_t)e * HID * (size_t)ld2;
  } else m0 = bx * 128;

  int tid = threadIdx.x, wid = tid >> 5, lane = tid & 31;

  for (int r = tid; r < 128; r += 256) {
    int rowid;
    if (gathered) { int idx = m0 + r; rowid = (idx < cnt_e) ? rows[(size_t)e * T_TOK + idx] : -1; }
    else rowid = m0 + r;
    s_rows[r] = rowid;
  }
  __syncthreads();

  v8f acc_g[4] = {}; v8f acc_u[4] = {};
  unsigned wf_off = (unsigned)(uintptr_t)(void*)s_wf;

  for (int kc = 0; kc < HID; kc += 64) {
    // stage activations (bf16, 128x64), 16B chunks
    for (int task = tid; task < 128 * 8; task += 256) {
      int r = task >> 3, c = (task & 7) << 3;
      int rowid = s_rows[r];
      if (rowid >= 0)
        *(uint4*)(s_a + r * AP + c) =
            *(const uint4*)(xb + (size_t)(rowid >> rshift) * HID + kc + c);
    }
    // gate weight tile via TDM (wave 0 only; TENSORcnt is per-wave)
    if (wid == 0) {
      tdm_load_2d(wf_off, w + (size_t)kc * ld2 + icol0, 64, 64, (int)ld2);
      __builtin_amdgcn_s_wait_tensorcnt(0);
    }
    __syncthreads();
    for (int i = tid; i < 64 * 64; i += 256) {      // cvt + transpose -> [n][k]
      int k = i >> 6, n = i & 63;
      s_wb[0][n * AP + k] = f32_to_bf16(s_wf[i]);
    }
    __syncthreads();
    if (wid == 0) {                                  // up weight tile
      tdm_load_2d(wf_off, w + (size_t)kc * ld2 + ic + icol0, 64, 64, (int)ld2);
      __builtin_amdgcn_s_wait_tensorcnt(0);
    }
    __syncthreads();
    for (int i = tid; i < 64 * 64; i += 256) {
      int k = i >> 6, n = i & 63;
      s_wb[1][n * AP + k] = f32_to_bf16(s_wf[i]);
    }
    __syncthreads();

    #pragma unroll
    for (int ks = 0; ks < 2; ++ks) {
      v16bf a = load_frag(s_a + wid * 16 * AP, AP, ks * 32, lane);
      #pragma unroll
      for (int nt = 0; nt < 4; ++nt) {
        v16bf bg = load_frag(&s_wb[0][nt * 16 * AP], AP, ks * 32, lane);
        v16bf bu = load_frag(&s_wb[1][nt * 16 * AP], AP, ks * 32, lane);
        acc_g[nt] = __builtin_amdgcn_wmma_f32_16x16x32_bf16(
            false, a, false, bg, (short)0, acc_g[nt], false, false);
        acc_u[nt] = __builtin_amdgcn_wmma_f32_16x16x32_bf16(
            false, a, false, bu, (short)0, acc_u[nt], false, false);
      }
    }
    __syncthreads();
  }

  int half = lane >> 4, nl = lane & 15;
  #pragma unroll
  for (int nt = 0; nt < 4; ++nt) {
    #pragma unroll
    for (int r = 0; r < 8; ++r) {
      float gv = acc_g[nt][r], uv = acc_u[nt][r];
      float h = (gv / (1.f + __expf(-gv))) * uv;    // silu(g) * u
      int m = wid * 16 + r + half * 8;
      int rowid = s_rows[m];
      if (rowid >= 0)
        inter[(size_t)rowid * ic + icol0 + nt * 16 + nl] = f32_to_bf16(h);
    }
  }
}

// ========================== kernel 3: gemm2 (down) ==========================
__global__ __launch_bounds__(256)
void gemm2_down_kernel(const unsigned short* __restrict__ inter,
                       const float* __restrict__ wdn,
                       const int* __restrict__ rows, const int* __restrict__ cnt,
                       const float* __restrict__ topw, float* __restrict__ out,
                       int ic, int gathered) {
  __shared__ int            s_rows[128];
  __shared__ float          s_wrow[128];
  __shared__ unsigned short s_a[128 * AP];
  __shared__ float          s_wf[64 * 64];
  __shared__ unsigned short s_wb[64 * AP];

  int bx = blockIdx.x, n0 = blockIdx.y * 64;
  int e = 0, m0, cnt_e = 0;
  const float* w = wdn;
  if (gathered) {
    e = bx >> 5; m0 = (bx & 31) * 128;
    cnt_e = cnt[e];
    if (m0 >= cnt_e) return;
    w += (size_t)e * (size_t)ic * HID;
  } else m0 = bx * 128;

  int tid = threadIdx.x, wid = tid >> 5, lane = tid & 31;

  for (int r = tid; r < 128; r += 256) {
    int rowid;
    if (gathered) { int idx = m0 + r; rowid = (idx < cnt_e) ? rows[(size_t)e * T_TOK + idx] : -1; }
    else rowid = m0 + r;
    s_rows[r] = rowid;
    s_wrow[r] = (gathered && rowid >= 0) ? topw[rowid] * 2.5f : 0.f;
  }
  __syncthreads();

  v8f acc[4] = {};
  unsigned wf_off = (unsigned)(uintptr_t)(void*)s_wf;

  for (int kc = 0; kc < ic; kc += 64) {
    for (int task = tid; task < 128 * 8; task += 256) {
      int r = task >> 3, c = (task & 7) << 3;
      int rowid = s_rows[r];
      if (rowid >= 0)
        *(uint4*)(s_a + r * AP + c) =
            *(const uint4*)(inter + (size_t)rowid * ic + kc + c);
    }
    if (wid == 0) {
      tdm_load_2d(wf_off, w + (size_t)kc * HID + n0, 64, 64, HID);
      __builtin_amdgcn_s_wait_tensorcnt(0);
    }
    __syncthreads();
    for (int i = tid; i < 64 * 64; i += 256) {
      int k = i >> 6, n = i & 63;
      s_wb[n * AP + k] = f32_to_bf16(s_wf[i]);
    }
    __syncthreads();

    #pragma unroll
    for (int ks = 0; ks < 2; ++ks) {
      v16bf a = load_frag(s_a + wid * 16 * AP, AP, ks * 32, lane);
      #pragma unroll
      for (int nt = 0; nt < 4; ++nt) {
        v16bf b = load_frag(&s_wb[nt * 16 * AP], AP, ks * 32, lane);
        acc[nt] = __builtin_amdgcn_wmma_f32_16x16x32_bf16(
            false, a, false, b, (short)0, acc[nt], false, false);
      }
    }
    __syncthreads();
  }

  int half = lane >> 4, nl = lane & 15;
  #pragma unroll
  for (int nt = 0; nt < 4; ++nt) {
    #pragma unroll
    for (int r = 0; r < 8; ++r) {
      int m = wid * 16 + r + half * 8;
      int rowid = s_rows[m];
      float v = acc[nt][r];
      int ncol = n0 + nt * 16 + nl;
      if (!gathered) {
        out[(size_t)rowid * HID + ncol] = v;               // shared: plain store
      } else if (rowid >= 0) {
        unsafeAtomicAdd(&out[(size_t)(rowid >> 3) * HID + ncol], s_wrow[m] * v);
      }
    }
  }
}

// =============================== launcher ===================================
extern "C" void kernel_launch(void* const* d_in, const int* in_sizes, int n_in,
                              void* d_out, int out_size, void* d_ws, size_t ws_size,
                              hipStream_t stream) {
  const float* x   = (const float*)d_in[0];
  const float* gw  = (const float*)d_in[1];
  const float* gb  = (const float*)d_in[2];
  const float* wgu = (const float*)d_in[3];
  const float* wdn = (const float*)d_in[4];
  const float* sgu = (const float*)d_in[5];
  const float* sdn = (const float*)d_in[6];
  float* out = (float*)d_out;

  char* ws = (char*)d_ws;
  size_t off = 0;
  unsigned short* xb = (unsigned short*)(ws + off); off += (size_t)T_TOK * HID * 2;
  float* topw        = (float*)(ws + off);          off += (size_t)T_TOK * TOPK * 4;
  int*   cnt         = (int*)(ws + off);            off += 128;
  int*   rows        = (int*)(ws + off);            off += (size_t)NEXP * T_TOK * 4;
  unsigned short* inter_r = (unsigned short*)(ws + off); off += (size_t)T_TOK * TOPK * INTER * 2;
  unsigned short* inter_s = (unsigned short*)(ws + off); off += (size_t)T_TOK * SIC * 2;

  (void)hipMemsetAsync(cnt, 0, 128, stream);

  int ncast = T_TOK * HID;
  cast_bf16_kernel<<<ncast / (256 * 4), 256, 0, stream>>>(x, xb, ncast);

  router_kernel<<<T_TOK / 8, 256, 0, stream>>>(x, gw, gb, topw, cnt, rows);

  // shared expert (writes d_out first)
  gemm1_silu_kernel<<<dim3(T_TOK / 128, SIC / 64), 256, 0, stream>>>(
      xb, sgu, nullptr, nullptr, inter_s, SIC, 0, 0);
  gemm2_down_kernel<<<dim3(T_TOK / 128, HID / 64), 256, 0, stream>>>(
      inter_s, sdn, nullptr, nullptr, nullptr, out, SIC, 0);

  // routed experts (atomic-accumulate into d_out, scaled by 2.5 * combine weight)
  gemm1_silu_kernel<<<dim3(NEXP * 32, INTER / 64), 256, 0, stream>>>(
      xb, wgu, rows, cnt, inter_r, INTER, 1, 3);
  gemm2_down_kernel<<<dim3(NEXP * 32, HID / 64), 256, 0, stream>>>(
      inter_r, wdn, rows, cnt, topw, out, INTER, 1);
}
